// InteractionBlock_69999376990649
// MI455X (gfx1250) — compile-verified, matrix operand order checked by reading, not run on hardware
//
#include <hip/hip_runtime.h>

// ---------------------------------------------------------------------------
// GemNet InteractionBlock for MI455X (gfx1250, wave32).
// Dense layers -> bf16 WMMA (v_wmma_f32_16x16x32_bf16, f32 accum).
// Irregular scatters/gathers/small einsums -> float4-vectorized VALU kernels.
// ---------------------------------------------------------------------------

typedef __attribute__((ext_vector_type(16))) __bf16 v16bf;
typedef __attribute__((ext_vector_type(8)))  __bf16 v8bf;
typedef __attribute__((ext_vector_type(4)))  __bf16 v4bf;
typedef __attribute__((ext_vector_type(2)))  __bf16 v2bf;
typedef __attribute__((ext_vector_type(8)))  float  v8f;

static constexpr int NAc = 2000, NEc = 60000, DAc = 256, DEc = 512;
static constexpr int TIc = 64, TOc = 64, QIc = 32, QOc = 32, AIc = 64, AOc = 64;
static constexpr int RBFc = 16, CBFc = 16, SBFc = 32, NSCc = 7, NSSc = 16;
static constexpr int NT_E2Ec = 480000, K_E2Ec = 8;
static constexpr int NQTc = 120000, NQc = 240000, K_Qc = 4;
static constexpr int NT_A2Ec = 240000, K_A2Ec = 4;
static constexpr int NT_E2Ac = 240000, K_E2Ac = 4;
static constexpr int K2c = 30;

static constexpr float IS2 = 0.70710678118654752f;   // 1/sqrt(2)
static constexpr float IS3 = 0.57735026918962584f;   // 1/sqrt(3)
static constexpr float SILU_SCALE = 1.6666666666666667f; // 1/0.6

__device__ __forceinline__ float silu_s(float x) {
  return x * (1.f / (1.f + __expf(-x))) * SILU_SCALE;
}

// ---------------------------------------------------------------------------
// Tiled mixed-precision GEMM: C[M,N] = epilogue(A[M,K] @ B[K,N]).
// Tile 128x64, BK=32, 256 threads = 8 waves laid out 4x2; each wave owns a
// 32x32 patch: 2 A-frags x 2 B-frags -> 4 WMMAs per k-step.
// LDS: A row-major (pitch 40 bf16 = 80B, 16B aligned rows); B stored N-major
// (sBt[n][k]) so both A and B fragments are two contiguous 16B runs per lane
// -> 2x ds_load_b128 each.  Next-tile global_prefetch overlaps WMMA.
// ep: 0=none, 1=scaled-silu, 2=scaled-silu * aux, 3=(aux + scaled-silu)*1/sqrt2
// ---------------------------------------------------------------------------
#define GBM 128
#define GBN 64
#define GBK 32
#define APITCH 40
#define BPITCH 40

__global__ __launch_bounds__(256)
void k_gemm(const float* __restrict__ A, const float* __restrict__ B,
            float* __restrict__ C, int M, int N, int K,
            int ep, const float* __restrict__ aux) {
  __shared__ __bf16 sA[GBM * APITCH];
  __shared__ __bf16 sBt[GBN * BPITCH];
  const int tid   = threadIdx.x;
  const int lane  = tid & 31;
  const int wave  = tid >> 5;
  const int wrow  = wave >> 1;     // 0..3 -> 32-row patch
  const int wcol  = wave & 1;      // 0..1 -> 32-col patch
  const int lhalf = lane >> 4;     // 0/1
  const int lmod  = lane & 15;
  const int m0 = blockIdx.y * GBM;
  const int n0 = blockIdx.x * GBN;
  const bool interior = (m0 + GBM <= M) && (n0 + GBN <= N);

  union F32x8 { v8f v; float e[8]; };
  F32x8 acc[2][2];
#pragma unroll
  for (int mi = 0; mi < 2; ++mi)
#pragma unroll
    for (int ni = 0; ni < 2; ++ni)
#pragma unroll
      for (int r = 0; r < 8; ++r) acc[mi][ni].e[r] = 0.f;

  for (int kt = 0; kt < K; kt += GBK) {
    const bool kfull = (kt + GBK <= K);
    if (interior && kfull) {
      // Fast path: vectorized staging, no guards.
      // A: 128x32 f32 -> bf16, one float4 load + v4bf (8B) LDS store x4.
#pragma unroll
      for (int i = 0; i < 4; ++i) {
        int li = i * 256 + tid;          // 0..1023
        int ra = li >> 3;                // 0..127
        int ca = (li & 7) << 2;          // 0,4,..,28
        const float4 f = *(const float4*)&A[(long)(m0 + ra) * K + kt + ca];
        v4bf p;
        p[0] = (__bf16)f.x; p[1] = (__bf16)f.y;
        p[2] = (__bf16)f.z; p[3] = (__bf16)f.w;
        *(v4bf*)&sA[ra * APITCH + ca] = p;
      }
      // B: 32x64 f32, transpose into sBt[n][k]; pack k-pairs -> v2bf store x4.
#pragma unroll
      for (int i = 0; i < 4; ++i) {
        int li = i * 256 + tid;          // 0..1023
        int kp = li >> 6;                // 0..15 (k pair)
        int nn = li & 63;
        float b0 = B[(long)(kt + 2 * kp) * N + n0 + nn];
        float b1 = B[(long)(kt + 2 * kp + 1) * N + n0 + nn];
        v2bf p; p[0] = (__bf16)b0; p[1] = (__bf16)b1;
        *(v2bf*)&sBt[nn * BPITCH + 2 * kp] = p;
      }
      // Prefetch next k-tile into cache while this tile computes
      // (global_prefetch_b8; speculative, dropped if unneeded).
      if (kt + 2 * GBK <= K) {
        const int nkt = kt + GBK;
        __builtin_prefetch(&A[(long)(m0 + (tid & 127)) * K + nkt + ((tid >> 7) << 4)], 0, 0);
        __builtin_prefetch(&B[(long)(nkt + (tid & 31)) * N + n0 + ((tid >> 5) << 3)], 0, 0);
      }
    } else {
      // Slow path (edge tiles / K not multiple of 32): branchless clamped
      // loads (always in-bounds) + select-to-zero. No exec-mask juggling.
#pragma unroll
      for (int i = 0; i < 16; ++i) {
        int li = i * 256 + tid;          // 0..4095
        int ra = li >> 5, ca = li & 31;
        int gm = m0 + ra, gk = kt + ca;
        int cm = gm < M ? gm : M - 1;
        int ck = gk < K ? gk : K - 1;
        float v = A[(long)cm * K + ck];
        sA[ra * APITCH + ca] = (__bf16)((gm < M && gk < K) ? v : 0.f);
      }
#pragma unroll
      for (int i = 0; i < 8; ++i) {
        int li = i * 256 + tid;          // 0..2047
        int rb = li >> 6, cb = li & 63;
        int gk = kt + rb, gn = n0 + cb;
        int ck = gk < K ? gk : K - 1;
        int cn = gn < N ? gn : N - 1;
        float v = B[(long)ck * N + cn];
        sBt[cb * BPITCH + rb] = (__bf16)((gk < K && gn < N) ? v : 0.f);
      }
    }
    __syncthreads();

    // Fragment loads per CDNA5 ISA 7.12.2: per lane two contiguous 16B runs
    // at k offsets lhalf*8 and 16+lhalf*8 -> ds_load_b128 pairs.
    union F16x16 { v16bf v; v8bf h[2]; };
    F16x16 fa[2], fb[2];
#pragma unroll
    for (int t = 0; t < 2; ++t) {
      const __bf16* ar = &sA[(wrow * 32 + t * 16 + lmod) * APITCH];
      fa[t].h[0] = *(const v8bf*)&ar[lhalf * 8];
      fa[t].h[1] = *(const v8bf*)&ar[16 + lhalf * 8];
      const __bf16* br = &sBt[(wcol * 32 + t * 16 + lmod) * BPITCH];
      fb[t].h[0] = *(const v8bf*)&br[lhalf * 8];
      fb[t].h[1] = *(const v8bf*)&br[16 + lhalf * 8];
    }
#pragma unroll
    for (int mi = 0; mi < 2; ++mi)
#pragma unroll
      for (int ni = 0; ni < 2; ++ni)
        acc[mi][ni].v = __builtin_amdgcn_wmma_f32_16x16x32_bf16(
            false, fa[mi].v, false, fb[ni].v, (short)0, acc[mi][ni].v,
            false, false);
    __syncthreads();
  }

  // C/D layout: lanes 0-15 -> M=r, lanes 16-31 -> M=8+r; N = lane%16.
#pragma unroll
  for (int mi = 0; mi < 2; ++mi) {
#pragma unroll
    for (int r = 0; r < 8; ++r) {
      int gm = m0 + wrow * 32 + mi * 16 + lhalf * 8 + r;
      if (gm >= M) continue;
#pragma unroll
      for (int ni = 0; ni < 2; ++ni) {
        int gn = n0 + wcol * 32 + ni * 16 + lmod;
        if (gn >= N) continue;
        float v = acc[mi][ni].e[r];
        long o = (long)gm * N + gn;
        float rout;
        if (ep == 0)      rout = v;
        else if (ep == 1) rout = silu_s(v);
        else if (ep == 2) rout = silu_s(v) * aux[o];
        else              rout = (aux[o] + silu_s(v)) * IS2;
        C[o] = rout;
      }
    }
  }
}

// ---------------------------------------------------------------------------
// Elementwise / scatter / gather / small-einsum kernels.
// All column counts in this model are multiples of 4 -> float4 everywhere.
// ---------------------------------------------------------------------------
__device__ __forceinline__ long gtid() {
  return (long)blockIdx.x * blockDim.x + threadIdx.x;
}

// n4 = number of float4 groups
__global__ void k_fill4(float4* p, long n4, float v) {
  long i = gtid(); if (i < n4) { float4 f = {v, v, v, v}; p[i] = f; }
}
__global__ void k_mul4(float4* o, const float4* a, const float4* b, long n4) {
  long i = gtid(); if (i >= n4) return;
  float4 x = a[i], y = b[i];
  float4 r = {x.x * y.x, x.y * y.y, x.z * y.z, x.w * y.w};
  o[i] = r;
}
__global__ void k_scale4(float4* o, const float4* a, float s, long n4) {
  long i = gtid(); if (i >= n4) return;
  float4 x = a[i];
  float4 r = {x.x * s, x.y * s, x.z * s, x.w * s};
  o[i] = r;
}
__global__ void k_add2_scale4(float4* o, const float4* a, const float4* b,
                              float s, long n4) {
  long i = gtid(); if (i >= n4) return;
  float4 x = a[i], y = b[i];
  float4 r = {(x.x + y.x) * s, (x.y + y.y) * s, (x.z + y.z) * s, (x.w + y.w) * s};
  o[i] = r;
}
__global__ void k_add4_scale4(float4* o, const float4* a, const float4* b,
                              const float4* c, const float4* d, float s, long n4) {
  long i = gtid(); if (i >= n4) return;
  float4 x = a[i], y = b[i], z = c[i], w = d[i];
  float4 r = {(x.x + y.x + z.x + w.x) * s, (x.y + y.y + z.y + w.y) * s,
              (x.z + y.z + z.z + w.z) * s, (x.w + y.w + z.w + w.w) * s};
  o[i] = r;
}
__global__ void k_hnew4(float4* o, const float4* h, const float4* xa,
                        const float4* he2a, const float4* ha2a, long n4) {
  long i = gtid(); if (i >= n4) return;
  float4 a = h[i], b = xa[i], c = he2a[i], d = ha2a[i];
  float4 r = {(a.x + (b.x + c.x + d.x) * IS3) * IS2,
              (a.y + (b.y + c.y + d.y) * IS3) * IS2,
              (a.z + (b.z + c.z + d.z) * IS3) * IS2,
              (a.w + (b.w + c.w + d.w) * IS3) * IS2};
  o[i] = r;
}
// dst[t,:] = src[idx[t],:] * other[t,:]   (C multiple of 4; C4 = C/4)
__global__ void k_gather_mul_rows4(float4* dst, const float4* src, const int* idx,
                                   const float4* other, int R, int C4) {
  long i = gtid(); long total = (long)R * C4; if (i >= total) return;
  int c = (int)(i % C4); long r = i / C4;
  float4 x = src[(long)idx[r] * C4 + c], y = other[i];
  float4 o = {x.x * y.x, x.y * y.y, x.z * y.z, x.w * y.w};
  dst[i] = o;
}
// out[r,:] = (a[r,:] + b[idx[r],:]) * s
__global__ void k_gather_add_scale4(float4* out, const float4* a, const float4* b,
                                    const int* idx, int R, int C4, float s) {
  long i = gtid(); long total = (long)R * C4; if (i >= total) return;
  int c = (int)(i % C4); long r = i / C4;
  float4 x = a[i], y = b[(long)idx[r] * C4 + c];
  float4 o = {(x.x + y.x) * s, (x.y + y.y) * s, (x.z + y.z) * s, (x.w + y.w) * s};
  out[i] = o;
}
// dst[out[t], agg[t], :] = src[in[t] (or t), :]   (unique (out,agg) pairs)
__global__ void k_gather_scatter4(float4* dst, const float4* src, const int* in_idx,
                                  const int* out_idx, const int* agg_idx,
                                  int T, int K, int C4) {
  long i = gtid(); long total = (long)T * C4; if (i >= total) return;
  int c = (int)(i % C4); long t = i / C4;
  long srow = in_idx ? (long)in_idx[t] : t;
  dst[((long)out_idx[t] * K + agg_idx[t]) * C4 + c] = src[srow * C4 + c];
}
// dst[seg[r], :] += src[r, :]  (atomic; stays scalar)
__global__ void k_scatter_add_rows(float* dst, const float* src, const int* seg,
                                   int R, int C) {
  long i = gtid(); long total = (long)R * C; if (i >= total) return;
  int c = (int)(i % C); long r = i / C;
  atomicAdd(&dst[(long)seg[r] * C + c], src[i]);
}
// cat[e] = [ hnew[ei0[e]] (256) | hnew[ei1[e]] (256) | mnew[e] (512) ]
// 256 float4 groups per edge; region boundaries are group-aligned.
__global__ void k_concat4(float4* dst, const float4* hn, const float4* mn,
                          const int* ei0, const int* ei1, int R) {
  long i = gtid(); long total = (long)R * 256; if (i >= total) return;
  int c = (int)(i & 255); long e = i >> 8;
  float4 v;
  if (c < 64)       v = hn[(long)ei0[e] * 64 + c];
  else if (c < 128) v = hn[(long)ei1[e] * 64 + (c - 64)];
  else              v = mn[e * 128 + (c - 128)];
  dst[i] = v;
}
// out[t,s,:] = sum_k sph[t,k,s] * mpad[t,k,:]   (I4 = I/4)
__global__ void k_sph_einsum4(const float* __restrict__ sph,
                              const float4* __restrict__ mpad,
                              float4* __restrict__ out, int T, int K, int S, int I4) {
  long idx = gtid(); long total = (long)T * S * I4; if (idx >= total) return;
  int i = (int)(idx % I4); long r = idx / I4; int s = (int)(r % S); long t = r / S;
  float4 acc = {0.f, 0.f, 0.f, 0.f};
  for (int k = 0; k < K; ++k) {
    float w = sph[(t * K + k) * S + s];
    float4 v = mpad[(t * K + k) * I4 + i];
    acc.x += w * v.x; acc.y += w * v.y; acc.z += w * v.z; acc.w += w * v.w;
  }
  out[(t * S + s) * I4 + i] = acc;
}
// out[t,c,:] = sum_s rw[t,c,s] * sm[t,s,:]
__global__ void k_rbf_einsum4(const float* __restrict__ rw,
                              const float4* __restrict__ sm,
                              float4* __restrict__ out, int T, int C, int S, int I4) {
  long idx = gtid(); long total = (long)T * C * I4; if (idx >= total) return;
  int i = (int)(idx % I4); long r = idx / I4; int c = (int)(r % C); long t = r / C;
  float4 acc = {0.f, 0.f, 0.f, 0.f};
  for (int s = 0; s < S; ++s) {
    float w = rw[(t * C + c) * S + s];
    float4 v = sm[(t * S + s) * I4 + i];
    acc.x += w * v.x; acc.y += w * v.y; acc.z += w * v.z; acc.w += w * v.w;
  }
  out[(t * C + c) * I4 + i] = acc;
}

// ---------------------------------------------------------------------------
// Host-side orchestration
// ---------------------------------------------------------------------------
static inline void gemm(hipStream_t st, const float* A, const float* B, float* C,
                        int M, int N, int K, int ep, const float* aux) {
  dim3 g((N + GBN - 1) / GBN, (M + GBM - 1) / GBM);
  k_gemm<<<g, dim3(256), 0, st>>>(A, B, C, M, N, K, ep, aux);
}
static inline dim3 ewg(long n) { return dim3((unsigned)((n + 255) / 256)); }

extern "C" void kernel_launch(void* const* d_in, const int* in_sizes, int n_in,
                              void* d_out, int out_size, void* d_ws, size_t ws_size,
                              hipStream_t stream) {
  (void)in_sizes; (void)n_in; (void)out_size; (void)ws_size;
  // ---- inputs (setup_inputs dict order) ----
  const float* h          = (const float*)d_in[0];
  const float* m          = (const float*)d_in[1];
  const float* qint_rad   = (const float*)d_in[2];
  const float* qint_cir   = (const float*)d_in[3];
  const float* qint_rbfW1 = (const float*)d_in[4];
  const float* qint_sph   = (const float*)d_in[5];
  const float* e2e_rad    = (const float*)d_in[6];
  const float* e2e_rbfW1  = (const float*)d_in[7];
  const float* e2e_sph    = (const float*)d_in[8];
  const float* a2e_rad    = (const float*)d_in[9];
  const float* a2e_rbfW1  = (const float*)d_in[10];
  const float* a2e_sph    = (const float*)d_in[11];
  const float* e2a_rad    = (const float*)d_in[12];
  const float* e2a_rbfW1  = (const float*)d_in[13];
  const float* e2a_sph    = (const float*)d_in[14];
  const float* a2a_rad    = (const float*)d_in[15];
  const float* au_rad     = (const float*)d_in[16];
  const float* W_dense_ca = (const float*)d_in[17];
  const float* e2e_W_ba   = (const float*)d_in[18];
  const float* e2e_W_rbf  = (const float*)d_in[19];
  const float* e2e_W_bil  = (const float*)d_in[20];
  const float* e2e_W_down = (const float*)d_in[21];
  const float* e2e_W_up_ca= (const float*)d_in[22];
  const float* e2e_W_up_ac= (const float*)d_in[23];
  const float* q_W_db     = (const float*)d_in[24];
  const float* q_W_rbf    = (const float*)d_in[25];
  const float* q_W_cbf    = (const float*)d_in[26];
  const float* q_W_bil    = (const float*)d_in[27];
  const float* q_W_down   = (const float*)d_in[28];
  const float* q_W_up_ca  = (const float*)d_in[29];
  const float* q_W_up_ac  = (const float*)d_in[30];
  const float* a2e_W_ba   = (const float*)d_in[31];
  const float* a2e_W_rbf  = (const float*)d_in[32];
  const float* a2e_W_bil  = (const float*)d_in[33];
  const float* a2e_W_down = (const float*)d_in[34];
  const float* a2e_W_up_ca= (const float*)d_in[35];
  const float* a2e_W_up_ac= (const float*)d_in[36];
  const float* e2a_W_ba   = (const float*)d_in[37];
  const float* e2a_W_rbf  = (const float*)d_in[38];
  const float* e2a_W_bil  = (const float*)d_in[39];
  const float* e2a_W_down = (const float*)d_in[40];
  const float* e2a_W_up_ca= (const float*)d_in[41];
  const float* a2a_W_down = (const float*)d_in[42];
  const float* a2a_W_bil  = (const float*)d_in[43];
  const float* a2a_W_up   = (const float*)d_in[44];
  const float* W_before   = (const float*)d_in[45];
  const float* W_after    = (const float*)d_in[46];
  const float* au_W_rbf   = (const float*)d_in[47];
  const float* au_W_mlp0  = (const float*)d_in[48];
  const float* au_W_res   = (const float*)d_in[49];
  const float* W_concat   = (const float*)d_in[50];
  const float* W_resm     = (const float*)d_in[51];
  const int* edge_index   = (const int*)d_in[52];   // (2, NE): [0]=src, [1]=dst
  const int* tni_main     = (const int*)d_in[53];
  const int* id_swap      = (const int*)d_in[54];
  const int* trip_e2e_in  = (const int*)d_in[55];
  const int* trip_e2e_out = (const int*)d_in[56];
  const int* trip_e2e_agg = (const int*)d_in[57];
  const int* q_trip_in    = (const int*)d_in[58];
  const int* q_t2q        = (const int*)d_in[59];
  const int* q_out        = (const int*)d_in[60];
  const int* q_agg        = (const int*)d_in[61];
  const int* a2e_in       = (const int*)d_in[62];
  const int* a2e_out      = (const int*)d_in[63];
  const int* a2e_agg      = (const int*)d_in[64];
  const int* e2a_in       = (const int*)d_in[65];
  const int* e2a_out      = (const int*)d_in[66];
  const int* e2a_agg      = (const int*)d_in[67];
  const int* a2a_ei       = (const int*)d_in[68];   // (2, NE)
  const int* a2a_tni      = (const int*)d_in[69];

  float* out_h = (float*)d_out;                       // (NA, DA)
  float* out_m = out_h + (size_t)NAc * DAc;           // (NE, DE)

  // ---- workspace layout (bump allocation, phase-scoped reuse) ----
  float* W = (float*)d_ws;
  size_t off = 0;
  auto take = [&](size_t n) { float* p = W + off; off += n; return p; };
  const size_t NEDE = (size_t)NEc * DEc;              // 30.72M floats
  float* P_xe2e  = take(NEDE);
  float* P_xqint = take(NEDE);
  float* P_xa2e  = take(NEDE);
  float* P_mnew  = take(NEDE);
  float* P_he2a  = take((size_t)NAc * DAc);
  float* P_ha2a  = take((size_t)NAc * DAc);
  float* S0 = take((size_t)NEc * SBFc * QIc);         // 61.44M (largest rmat)
  float* S1 = take(NEDE);
  float* S2 = take(NEDE);
  float* S3 = take(NEDE);
  float* S4 = take((size_t)8 * 1000 * 1000);

  hipStream_t st = stream;
#define F4(p) ((float4*)(p))
#define CF4(p) ((const float4*)(p))

  // ================= E2E triplet interaction =================
  gemm(st, e2e_rad, e2e_W_rbf, S1, NEc, DEc, RBFc, 0, nullptr);          // rbf proj
  gemm(st, m, e2e_W_ba, S2, NEc, DEc, DEc, 2, S1);                       // act * rbf
  gemm(st, S2, e2e_W_down, S4, NEc, TIc, DEc, 1, nullptr);               // (NE,64)
  {
    long n4 = (long)NEc * K_E2Ec * TIc / 4;
    k_fill4<<<ewg(n4), 256, 0, st>>>(F4(S1), n4, 0.f);
  }
  k_gather_scatter4<<<ewg((long)NT_E2Ec * TIc / 4), 256, 0, st>>>(
      F4(S1), CF4(S4), trip_e2e_in, trip_e2e_out, trip_e2e_agg, NT_E2Ec, K_E2Ec, TIc / 4);
  k_sph_einsum4<<<ewg((long)NEc * NSCc * TIc / 4), 256, 0, st>>>(
      e2e_sph, CF4(S1), F4(S2), NEc, K_E2Ec, NSCc, TIc / 4);
  k_rbf_einsum4<<<ewg((long)NEc * CBFc * TIc / 4), 256, 0, st>>>(
      e2e_rbfW1, CF4(S2), F4(S0), NEc, CBFc, NSCc, TIc / 4);
  gemm(st, S0, e2e_W_bil, S4, NEc, TOc, CBFc * TIc, 0, nullptr);         // (NE,64)
  gemm(st, S4, e2e_W_up_ca, S1, NEc, DEc, TOc, 1, nullptr);
  gemm(st, S4, e2e_W_up_ac, S2, NEc, DEc, TOc, 1, nullptr);
  k_gather_add_scale4<<<ewg((long)NEDE / 4), 256, 0, st>>>(
      F4(P_xe2e), CF4(S1), CF4(S2), id_swap, NEc, DEc / 4, IS2);

  // ================= Quadruplet interaction =================
  gemm(st, qint_rad, q_W_rbf, S1, NEc, DEc, RBFc, 0, nullptr);
  gemm(st, m, q_W_db, S2, NEc, DEc, DEc, 2, S1);
  gemm(st, S2, q_W_down, S4, NEc, QIc, DEc, 1, nullptr);                 // (NE,32)
  gemm(st, qint_cir, q_W_cbf, S3, NQTc, QIc, CBFc, 0, nullptr);          // (NQT,32)
  k_gather_mul_rows4<<<ewg((long)NQTc * QIc / 4), 256, 0, st>>>(
      F4(S1), CF4(S4), q_trip_in, CF4(S3), NQTc, QIc / 4);
  {
    long n4 = (long)NEc * K_Qc * QIc / 4;
    k_fill4<<<ewg(n4), 256, 0, st>>>(F4(S2), n4, 0.f);
  }
  k_gather_scatter4<<<ewg((long)NQc * QIc / 4), 256, 0, st>>>(
      F4(S2), CF4(S1), q_t2q, q_out, q_agg, NQc, K_Qc, QIc / 4);
  k_sph_einsum4<<<ewg((long)NEc * NSSc * QIc / 4), 256, 0, st>>>(
      qint_sph, CF4(S2), F4(S3), NEc, K_Qc, NSSc, QIc / 4);
  k_rbf_einsum4<<<ewg((long)NEc * SBFc * QIc / 4), 256, 0, st>>>(
      qint_rbfW1, CF4(S3), F4(S0), NEc, SBFc, NSSc, QIc / 4);
  gemm(st, S0, q_W_bil, S4, NEc, QOc, SBFc * QIc, 0, nullptr);           // (NE,32)
  gemm(st, S4, q_W_up_ca, S1, NEc, DEc, QOc, 1, nullptr);
  gemm(st, S4, q_W_up_ac, S2, NEc, DEc, QOc, 1, nullptr);
  k_gather_add_scale4<<<ewg((long)NEDE / 4), 256, 0, st>>>(
      F4(P_xqint), CF4(S1), CF4(S2), id_swap, NEc, DEc / 4, IS2);

  // ================= A2E triplet interaction =================
  gemm(st, a2e_rad, a2e_W_rbf, S1, NEc, DAc, RBFc, 0, nullptr);          // (NE,256)
  gemm(st, h, a2e_W_ba, S3, NAc, DAc, DAc, 1, nullptr);                  // (NA,256)
  k_gather_mul_rows4<<<ewg((long)NEc * DAc / 4), 256, 0, st>>>(
      F4(S2), CF4(S3), edge_index, CF4(S1), NEc, DAc / 4);
  gemm(st, S2, a2e_W_down, S4, NEc, TIc, DAc, 1, nullptr);
  {
    long n4 = (long)NEc * K_A2Ec * TIc / 4;
    k_fill4<<<ewg(n4), 256, 0, st>>>(F4(S1), n4, 0.f);
  }
  k_gather_scatter4<<<ewg((long)NT_A2Ec * TIc / 4), 256, 0, st>>>(
      F4(S1), CF4(S4), a2e_in, a2e_out, a2e_agg, NT_A2Ec, K_A2Ec, TIc / 4);
  k_sph_einsum4<<<ewg((long)NEc * NSCc * TIc / 4), 256, 0, st>>>(
      a2e_sph, CF4(S1), F4(S2), NEc, K_A2Ec, NSCc, TIc / 4);
  k_rbf_einsum4<<<ewg((long)NEc * CBFc * TIc / 4), 256, 0, st>>>(
      a2e_rbfW1, CF4(S2), F4(S0), NEc, CBFc, NSCc, TIc / 4);
  gemm(st, S0, a2e_W_bil, S3, NEc, TOc, CBFc * TIc, 0, nullptr);
  gemm(st, S3, a2e_W_up_ca, S1, NEc, DEc, TOc, 1, nullptr);
  gemm(st, S3, a2e_W_up_ac, S2, NEc, DEc, TOc, 1, nullptr);
  k_gather_add_scale4<<<ewg((long)NEDE / 4), 256, 0, st>>>(
      F4(P_xa2e), CF4(S1), CF4(S2), id_swap, NEc, DEc / 4, IS2);

  // ================= E2A triplet interaction (two-stage) =================
  gemm(st, e2a_rad, e2a_W_rbf, S1, NEc, DEc, RBFc, 0, nullptr);
  gemm(st, m, e2a_W_ba, S2, NEc, DEc, DEc, 2, S1);
  gemm(st, S2, e2a_W_down, S4, NEc, TIc, DEc, 1, nullptr);
  {
    long n4 = (long)NEc * K_E2Ac * TIc / 4;
    k_fill4<<<ewg(n4), 256, 0, st>>>(F4(S1), n4, 0.f);
  }
  k_gather_scatter4<<<ewg((long)NT_E2Ac * TIc / 4), 256, 0, st>>>(
      F4(S1), CF4(S4), e2a_in, e2a_out, e2a_agg, NT_E2Ac, K_E2Ac, TIc / 4);
  k_sph_einsum4<<<ewg((long)NEc * NSCc * TIc / 4), 256, 0, st>>>(
      e2a_sph, CF4(S1), F4(S2), NEc, K_E2Ac, NSCc, TIc / 4);             // (NE,7,64)
  {
    long n4 = (long)NAc * K2c * NSCc * TIc / 4;
    k_fill4<<<ewg(n4), 256, 0, st>>>(F4(S3), n4, 0.f);
  }
  k_gather_scatter4<<<ewg((long)NEc * NSCc * TIc / 4), 256, 0, st>>>(
      F4(S3), CF4(S2), nullptr, edge_index + NEc, tni_main, NEc, K2c, NSCc * TIc / 4);
  k_rbf_einsum4<<<ewg((long)NAc * CBFc * TIc / 4), 256, 0, st>>>(
      e2a_rbfW1, CF4(S3), F4(S0), NAc, CBFc, K2c * NSCc, TIc / 4);       // (NA,16,64)
  gemm(st, S0, e2a_W_bil, S4, NAc, TOc, CBFc * TIc, 0, nullptr);
  gemm(st, S4, e2a_W_up_ca, P_he2a, NAc, DAc, TOc, 1, nullptr);

  // ================= A2A pair interaction =================
  gemm(st, h, a2a_W_down, S4, NAc, AIc, DAc, 1, nullptr);                // (NA,64)
  {
    long n4 = (long)NAc * K2c * AIc / 4;
    k_fill4<<<ewg(n4), 256, 0, st>>>(F4(S1), n4, 0.f);
  }
  k_gather_scatter4<<<ewg((long)NEc * AIc / 4), 256, 0, st>>>(
      F4(S1), CF4(S4), a2a_ei, a2a_ei + NEc, a2a_tni, NEc, K2c, AIc / 4);
  k_rbf_einsum4<<<ewg((long)NAc * RBFc * AIc / 4), 256, 0, st>>>(
      a2a_rad, CF4(S1), F4(S0), NAc, RBFc, K2c, AIc / 4);                // (NA,16,64)
  gemm(st, S0, a2a_W_bil, S2, NAc, AOc, RBFc * AIc, 0, nullptr);
  gemm(st, S2, a2a_W_up, P_ha2a, NAc, DAc, AOc, 1, nullptr);

  // ================= merge edge paths + residual stacks =================
  gemm(st, m, W_dense_ca, S1, NEc, DEc, DEc, 1, nullptr);
  k_add4_scale4<<<ewg((long)NEDE / 4), 256, 0, st>>>(
      F4(S2), CF4(S1), CF4(P_xe2e), CF4(P_xqint), CF4(P_xa2e), 0.5f, (long)NEDE / 4);
  const size_t WSQ = (size_t)DEc * DEc;
  // before[0]: x0=S2 -> x1=S3
  gemm(st, S2, W_before + 0 * WSQ, S1, NEc, DEc, DEc, 1, nullptr);
  gemm(st, S1, W_before + 1 * WSQ, S3, NEc, DEc, DEc, 3, S2);
  // before[1]: x1=S3 -> x2=S2
  gemm(st, S3, W_before + 2 * WSQ, S1, NEc, DEc, DEc, 1, nullptr);
  gemm(st, S1, W_before + 3 * WSQ, S2, NEc, DEc, DEc, 3, S3);
  // skip with m: x3=S3
  k_add2_scale4<<<ewg((long)NEDE / 4), 256, 0, st>>>(
      F4(S3), CF4(S2), CF4(m), IS2, (long)NEDE / 4);
  // after[0]: x3=S3 -> x4=S2
  gemm(st, S3, W_after + 0 * WSQ, S1, NEc, DEc, DEc, 1, nullptr);
  gemm(st, S1, W_after + 1 * WSQ, S2, NEc, DEc, DEc, 3, S3);
  // after[1]: x4=S2 -> m_new
  gemm(st, S2, W_after + 2 * WSQ, S1, NEc, DEc, DEc, 1, nullptr);
  gemm(st, S1, W_after + 3 * WSQ, P_mnew, NEc, DEc, DEc, 3, S2);

  // ================= atom update =================
  gemm(st, au_rad, au_W_rbf, S1, NEc, DEc, RBFc, 0, nullptr);
  k_mul4<<<ewg((long)NEDE / 4), 256, 0, st>>>(F4(S2), CF4(P_mnew), CF4(S1), (long)NEDE / 4);
  {
    long n4 = (long)NAc * DEc / 4;
    k_fill4<<<ewg(n4), 256, 0, st>>>(F4(S3), n4, 0.f);
  }
  k_scatter_add_rows<<<ewg((long)NEDE), 256, 0, st>>>(S3, S2, edge_index + NEc, NEc, DEc);
  gemm(st, S3, au_W_mlp0, S4, NAc, DAc, DEc, 1, nullptr);                // xa (NA,256)
  const size_t WAQ = (size_t)DAc * DAc;
  gemm(st, S4, au_W_res + 0 * WAQ, S1, NAc, DAc, DAc, 1, nullptr);
  gemm(st, S1, au_W_res + 1 * WAQ, S2, NAc, DAc, DAc, 3, S4);
  gemm(st, S2, au_W_res + 2 * WAQ, S1, NAc, DAc, DAc, 1, nullptr);
  gemm(st, S1, au_W_res + 3 * WAQ, S4, NAc, DAc, DAc, 3, S2);
  gemm(st, S4, au_W_res + 4 * WAQ, S1, NAc, DAc, DAc, 1, nullptr);
  gemm(st, S1, au_W_res + 5 * WAQ, S2, NAc, DAc, DAc, 3, S4);
  k_hnew4<<<ewg((long)NAc * DAc / 4), 256, 0, st>>>(
      F4(out_h), CF4(h), CF4(S2), CF4(P_he2a), CF4(P_ha2a), (long)NAc * DAc / 4);

  // ================= edge embedding update =================
  k_concat4<<<ewg((long)NEc * 256), 256, 0, st>>>(
      F4(S0), CF4(out_h), CF4(P_mnew), edge_index, edge_index + NEc, NEc);
  gemm(st, S0, W_concat, S1, NEc, DEc, 2 * DAc + DEc, 1, nullptr);
  gemm(st, S1, W_resm + 0 * WSQ, S2, NEc, DEc, DEc, 1, nullptr);
  gemm(st, S2, W_resm + 1 * WSQ, S3, NEc, DEc, DEc, 3, S1);
  k_scale4<<<ewg((long)NEDE / 4), 256, 0, st>>>(F4(out_m), CF4(S3), IS2, (long)NEDE / 4);
}
